// LoRALinear_48524540510754
// MI455X (gfx1250) — matrix operand order repaired and checked by reading, not compiled, not run on hardware
//
#include <hip/hip_runtime.h>

// ---------------------------------------------------------------------------
// LoRA forward for MI455X (gfx1250, wave32, WMMA).
//   out[T,OUT] = (dropout(x) @ A^T) @ B^T * SCALING
// T=16384, IN=4096, OUT=4096, R=64, SCALING = 128/64 = 2.0 (exact in bf16).
// Memory-bound: ~805 MB of HBM traffic vs 17.2 GFLOP -> bf16 WMMA, f32 accum,
// one streaming pass over x/drop_u and out. Phase-1 K-loop is software-
// pipelined (double-buffered register loads) so each wave keeps a full chunk
// of loads in flight instead of stalling on s_wait_loadcnt 0 per WMMA.
// ---------------------------------------------------------------------------

typedef __attribute__((ext_vector_type(16))) __bf16 v16bf;
typedef __attribute__((ext_vector_type(8)))  __bf16 v8bf;
typedef __attribute__((ext_vector_type(8)))  float  v8f;
typedef __attribute__((ext_vector_type(4)))  float  v4f;

#define T_DIM   16384
#define IN_DIM  4096
#define OUT_DIM 4096
#define R_DIM   64
#define PDROP   0.1f
#define KEEPSC  (1.0f / 0.9f)

// ---------------------------------------------------------------------------
// Pack lora_A (and 2.0*lora_B) into bf16 WMMA B-fragment layout:
//   frag[(chunk*ntiles + ntile)*32 + lane][e], e = 0..15 halves per lane.
// B-matrix layout (32x16, 16-bit): lane&15 -> N, (lane>>4, e) -> K in table order.
// Each lane's fragment is 16 contiguous bf16 = 32 bytes -> coalesced b128 x2.
// ---------------------------------------------------------------------------
__global__ void __launch_bounds__(256)
lora_pack(const float* __restrict__ A, const float* __restrict__ B,
          __bf16* __restrict__ Apk, __bf16* __restrict__ Bpk) {
    const int f = blockIdx.x * blockDim.x + threadIdx.x;
    const int NA = R_DIM * IN_DIM;               // 262144 elements
    if (f < NA) {
        const int e    = f & 15;
        const int lane = (f >> 4) & 31;
        const int rt   = (f >> 9) & 3;
        const int kc   = f >> 11;
        Apk[f] = (__bf16)A[(size_t)(rt * 16 + (lane & 15)) * IN_DIM + kc * 32 + (e & 7) + 8 * (lane >> 4) + 16 * (e >> 3)];
    } else {
        const int g    = f - NA;                 // 262144 elements
        const int e    = g & 15;
        const int lane = (g >> 4) & 31;
        const int ot   = (g >> 9) & 255;
        const int kc   = g >> 17;                // 0..1
        // fold SCALING = 2.0 (exact) into packed lora_B
        Bpk[g] = (__bf16)(2.0f * B[(size_t)(ot * 16 + (lane & 15)) * R_DIM + kc * 32 + (e & 7) + 8 * (lane >> 4) + 16 * (e >> 3)]);
    }
}

// ---------------------------------------------------------------------------
// Phase 1: h[T,64] = dropout(x) @ lora_A^T  (bf16 inputs, f32 accum).
// Block = 128 threads (4 waves); each wave computes a 16(t) x 64(r) strip.
// Double-buffered software pipeline: loads for chunk kc+1 issue before the
// WMMAs of chunk kc, so waits are partial (loadcnt <= 16) instead of 0.
// ---------------------------------------------------------------------------
__global__ void __launch_bounds__(128)
lora_phase1(const float* __restrict__ x, const float* __restrict__ u,
            const __bf16* __restrict__ Apk, __bf16* __restrict__ hbf) {
    const int lane = threadIdx.x & 31;
    const int wave = threadIdx.x >> 5;
    const int tb   = blockIdx.x * 64 + wave * 16;
    const int m    = lane & 15;
    const int hi   = lane >> 4;

    const float*  xrow  = x + (size_t)(tb + m) * IN_DIM + hi * 8;
    const float*  urow  = u + (size_t)(tb + m) * IN_DIM + hi * 8;
    const __bf16* bbase = Apk + lane * 16;

    v8f c0 = {}, c1 = {}, c2 = {}, c3 = {};

    v4f   xa[4], ua[4], xb[4], ub[4];
    v16bf pa0, pa1, pa2, pa3, pb0, pb1, pb2, pb3;

#define LOAD_XU(kc, XQ, UQ) do {                                              \
        const float* _xp = xrow + (size_t)(kc) * 32;                          \
        const float* _up = urow + (size_t)(kc) * 32;                          \
        XQ[0] = *(const v4f*)(_xp);      XQ[1] = *(const v4f*)(_xp + 4);      \
        XQ[2] = *(const v4f*)(_xp + 16); XQ[3] = *(const v4f*)(_xp + 20);     \
        UQ[0] = *(const v4f*)(_up);      UQ[1] = *(const v4f*)(_up + 4);      \
        UQ[2] = *(const v4f*)(_up + 16); UQ[3] = *(const v4f*)(_up + 20);     \
    } while (0)

#define LOAD_B(kc, B0, B1, B2, B3) do {                                       \
        const __bf16* _bp = bbase + (size_t)(kc) * 2048;                      \
        B0 = *(const v16bf*)(_bp);                                            \
        B1 = *(const v16bf*)(_bp + 512);                                      \
        B2 = *(const v16bf*)(_bp + 1024);                                     \
        B3 = *(const v16bf*)(_bp + 1536);                                     \
    } while (0)

#define COMPUTE(XQ, UQ, B0, B1, B2, B3) do {                                  \
        const float* _xv = (const float*)(XQ);                                \
        const float* _uv = (const float*)(UQ);                                \
        v16bf _a;                                                             \
        _Pragma("unroll")                                                     \
        for (int _i = 0; _i < 16; ++_i) {                                     \
            const float _k = (_uv[_i] >= PDROP) ? KEEPSC : 0.0f;              \
            _a[_i] = (__bf16)(_xv[_i] * _k);                                  \
        }                                                                     \
        c0 = __builtin_amdgcn_wmma_f32_16x16x32_bf16(false, _a, false, B0, (short)0, c0, false, false); \
        c1 = __builtin_amdgcn_wmma_f32_16x16x32_bf16(false, _a, false, B1, (short)0, c1, false, false); \
        c2 = __builtin_amdgcn_wmma_f32_16x16x32_bf16(false, _a, false, B2, (short)0, c2, false, false); \
        c3 = __builtin_amdgcn_wmma_f32_16x16x32_bf16(false, _a, false, B3, (short)0, c3, false, false); \
    } while (0)

    // prologue: fill buffer A with chunk 0
    LOAD_XU(0, xa, ua);
    LOAD_B(0, pa0, pa1, pa2, pa3);

    // 2x-unrolled steady state: constant buffer indices -> no scratch spills
    for (int kc = 0; kc < IN_DIM / 32; kc += 2) {
        LOAD_XU(kc + 1, xb, ub);
        LOAD_B(kc + 1, pb0, pb1, pb2, pb3);
        COMPUTE(xa, ua, pa0, pa1, pa2, pa3);

        const int kn = (kc + 2 < IN_DIM / 32) ? (kc + 2) : (kc + 1); // clamp (dup load, harmless)
        LOAD_XU(kn, xa, ua);
        LOAD_B(kn, pa0, pa1, pa2, pa3);
        COMPUTE(xb, ub, pb0, pb1, pb2, pb3);
    }

    // C/D layout: VGPR j -> row m = j + 8*(lane>>4), col n = lane&15
#pragma unroll
    for (int j = 0; j < 8; ++j) {
        const int row = tb + j + hi * 8;
        __bf16* hp = hbf + (size_t)row * R_DIM + m;
        hp[0]  = (__bf16)c0[j];
        hp[16] = (__bf16)c1[j];
        hp[32] = (__bf16)c2[j];
        hp[48] = (__bf16)c3[j];
    }
#undef LOAD_XU
#undef LOAD_B
#undef COMPUTE
}

// ---------------------------------------------------------------------------
// Phase 2: out[T,OUT] = h @ (2*lora_B)^T  (f32 out; scaling pre-folded).
// Block = 256 threads (8 waves); wave w -> 16(t) x 64(o) output tile.
// K=64: hoist ALL fragment loads to the top (one load clause), then 8
// back-to-back WMMAs, then coalesced f32 stores. Store-bound by design.
// ---------------------------------------------------------------------------
__global__ void __launch_bounds__(256)
lora_phase2(const __bf16* __restrict__ hbf, const __bf16* __restrict__ Bpk,
            float* __restrict__ out) {
    const int lane = threadIdx.x & 31;
    const int wave = threadIdx.x >> 5;
    const int tb   = blockIdx.y * 128 + wave * 16;
    const int ob   = blockIdx.x * 64;
    const int m    = lane & 15;
    const int hi   = lane >> 4;
    const int k0   = hi * 8;

    // A-fragment raw data for both K-chunks (4 x b128)
    const __bf16* hrow = hbf + (size_t)(tb + m) * R_DIM;
    const v8bf a0lo = *(const v8bf*)(hrow + k0);
    const v8bf a0hi = *(const v8bf*)(hrow + k0 + 16);
    const v8bf a1lo = *(const v8bf*)(hrow + 32 + k0);
    const v8bf a1hi = *(const v8bf*)(hrow + 32 + k0 + 16);

    // B-fragments for both K-chunks x 4 o-tiles (8 x 32B)
    const __bf16* bp  = Bpk + (size_t)(ob >> 4) * 512 + lane * 16;
    const __bf16* bp1 = bp + (size_t)256 * 512;
    const v16bf b00 = *(const v16bf*)(bp);
    const v16bf b01 = *(const v16bf*)(bp + 512);
    const v16bf b02 = *(const v16bf*)(bp + 1024);
    const v16bf b03 = *(const v16bf*)(bp + 1536);
    const v16bf b10 = *(const v16bf*)(bp1);
    const v16bf b11 = *(const v16bf*)(bp1 + 512);
    const v16bf b12 = *(const v16bf*)(bp1 + 1024);
    const v16bf b13 = *(const v16bf*)(bp1 + 1536);

    v16bf a0, a1;
#pragma unroll
    for (int i = 0; i < 8; ++i) {
        a0[i] = a0lo[i];  a0[8 + i] = a0hi[i];
        a1[i] = a1lo[i];  a1[8 + i] = a1hi[i];
    }

    v8f c0 = {}, c1 = {}, c2 = {}, c3 = {};
    c0 = __builtin_amdgcn_wmma_f32_16x16x32_bf16(false, a0, false, b00, (short)0, c0, false, false);
    c1 = __builtin_amdgcn_wmma_f32_16x16x32_bf16(false, a0, false, b01, (short)0, c1, false, false);
    c2 = __builtin_amdgcn_wmma_f32_16x16x32_bf16(false, a0, false, b02, (short)0, c2, false, false);
    c3 = __builtin_amdgcn_wmma_f32_16x16x32_bf16(false, a0, false, b03, (short)0, c3, false, false);
    c0 = __builtin_amdgcn_wmma_f32_16x16x32_bf16(false, a1, false, b10, (short)0, c0, false, false);
    c1 = __builtin_amdgcn_wmma_f32_16x16x32_bf16(false, a1, false, b11, (short)0, c1, false, false);
    c2 = __builtin_amdgcn_wmma_f32_16x16x32_bf16(false, a1, false, b12, (short)0, c2, false, false);
    c3 = __builtin_amdgcn_wmma_f32_16x16x32_bf16(false, a1, false, b13, (short)0, c3, false, false);

#pragma unroll
    for (int j = 0; j < 8; ++j) {
        const int row = tb + j + hi * 8;
        float* op = out + (size_t)row * OUT_DIM + ob + m;
        op[0]  = c0[j];
        op[16] = c1[j];
        op[32] = c2[j];
        op[48] = c3[j];
    }
}

// ---------------------------------------------------------------------------
extern "C" void kernel_launch(void* const* d_in, const int* in_sizes, int n_in,
                              void* d_out, int out_size, void* d_ws, size_t ws_size,
                              hipStream_t stream) {
    (void)in_sizes; (void)n_in; (void)out_size; (void)ws_size;
    const float* x = (const float*)d_in[0];   // [T, IN]
    const float* A = (const float*)d_in[1];   // [R, IN]
    const float* B = (const float*)d_in[2];   // [OUT, R]
    const float* u = (const float*)d_in[3];   // [T, IN]
    float* out = (float*)d_out;               // [T, OUT]

    char* ws = (char*)d_ws;
    __bf16* Apk = (__bf16*)(ws);                     // 512 KB
    __bf16* Bpk = (__bf16*)(ws + (512 << 10));       // 512 KB
    __bf16* hbf = (__bf16*)(ws + (1024 << 10));      // 2 MB

    // 1) pack weights into WMMA fragment order (bf16, scaling folded into B)
    lora_pack<<<(2 * R_DIM * IN_DIM) / 256, 256, 0, stream>>>(A, B, Apk, Bpk);

    // 2) h = dropout(x) @ A^T   (reads 536 MB, writes 2 MB)
    lora_phase1<<<T_DIM / 64, 128, 0, stream>>>(x, u, Apk, hbf);

    // 3) out = h @ (2B)^T       (writes 268 MB)
    dim3 g2(OUT_DIM / 64, T_DIM / 128);
    lora_phase2<<<g2, 256, 0, stream>>>(hbf, Bpk, out);
}
